// GCNBlock_7129645711554
// MI455X (gfx1250) — matrix operand order repaired and checked by reading, not compiled, not run on hardware
//
#include <hip/hip_runtime.h>

typedef __attribute__((ext_vector_type(16))) _Float16 v16h;
typedef __attribute__((ext_vector_type(8)))  float    v8f;

#define D 64

// ---------------------------------------------------------------------------
// WMMA helper: D = A(16x32 f16) * B(32x16 f16) + C(16x16 f32)
// ---------------------------------------------------------------------------
__device__ __forceinline__ v8f wmma16(v16h a, v16h b, v8f c) {
    return __builtin_amdgcn_wmma_f32_16x16x32_f16(
        /*neg_a=*/false, a, /*neg_b=*/false, b,
        /*c_mod=*/(short)0, c, /*reuse_a=*/false, /*reuse_b=*/false);
}

// Build a 16-bit A fragment for one K-chunk of 32.
// 16-bit A 16x32 layout: lane half h=lane/16, M=lane%16.
//   elements 0..7  -> K = k0 + h*8 + e
//   elements 8..15 -> K = k0 + 16 + h*8 + (e-8)
// rowPtr points at the f32 source row (256B aligned); scale applied pre-cvt.
__device__ __forceinline__ v16h load_a_frag(const float* rowPtr, int k0, int h, float s) {
    const float* p = rowPtr + k0 + h * 8;
    float4 f0 = *(const float4*)(p);
    float4 f1 = *(const float4*)(p + 4);
    float4 f2 = *(const float4*)(p + 16);
    float4 f3 = *(const float4*)(p + 20);
    v16h a;
    a[0]  = (_Float16)(f0.x * s); a[1]  = (_Float16)(f0.y * s);
    a[2]  = (_Float16)(f0.z * s); a[3]  = (_Float16)(f0.w * s);
    a[4]  = (_Float16)(f1.x * s); a[5]  = (_Float16)(f1.y * s);
    a[6]  = (_Float16)(f1.z * s); a[7]  = (_Float16)(f1.w * s);
    a[8]  = (_Float16)(f2.x * s); a[9]  = (_Float16)(f2.y * s);
    a[10] = (_Float16)(f2.z * s); a[11] = (_Float16)(f2.w * s);
    a[12] = (_Float16)(f3.x * s); a[13] = (_Float16)(f3.y * s);
    a[14] = (_Float16)(f3.z * s); a[15] = (_Float16)(f3.w * s);
    return a;
}

// ---------------------------------------------------------------------------
// Workspace init
// ---------------------------------------------------------------------------
__global__ void zero_kernel(float* p, int n) {
    int i = blockIdx.x * blockDim.x + threadIdx.x;
    if (i < n) p[i] = 0.0f;
}

// Repack a (Kdim x 64) f32 weight matrix into WMMA B-fragment order (f16).
// B 16-bit layout per 32-K chunk: lane half h, N = nt*16 + lane%16,
//   element e -> K = chunk*32 + h*8 + (e&7) + (e>>3)*16   (matches A table order).
// frag index = (((chunk*4 + nt)*32 + lane)*16 + e)
__global__ void pack_kernel(const float* Mat, _Float16* frag, int Kdim) {
    int tid = blockIdx.x * blockDim.x + threadIdx.x;
    int total = Kdim * D;
    if (tid >= total) return;
    int e     = tid & 15;
    int lane  = (tid >> 4) & 31;
    int nt    = (tid >> 9) & 3;
    int chunk = tid >> 11;
    int K = chunk * 32 + (lane >> 4) * 8 + (e & 7) + (e >> 3) * 16;
    int n = nt * 16 + (lane & 15);
    frag[tid] = (_Float16)Mat[K * D + n];
}

// ---------------------------------------------------------------------------
// Phase 1: per-edge messages (both directions), relu, atomic scatter-sum
// One wave = 16 edges. 32 WMMAs per wave (edge_emb K-chunks shared).
// ---------------------------------------------------------------------------
__global__ void msg_kernel(const float* __restrict__ node_emb,
                           const float* __restrict__ edge_emb,
                           const float* __restrict__ feat_emb,
                           const int* __restrict__ src,
                           const int* __restrict__ dst,
                           const _Float16* __restrict__ fragP,
                           const float* __restrict__ bP,
                           float* __restrict__ msgSum,
                           float* __restrict__ msgfSum,
                           float* __restrict__ cntSrc,
                           float* __restrict__ cntDst,
                           int nEdges)
{
    int wave = (blockIdx.x * blockDim.x + threadIdx.x) >> 5;
    int lane = threadIdx.x & 31;
    int e0 = wave * 16;
    if (e0 >= nEdges) return;          // wave-uniform: EXEC all-1s past here
    int h = lane >> 4;
    int m = lane & 15;
    int e = e0 + m;

    int si = src[e];
    int di = dst[e];
    if (h == 0) {                      // one count per edge
        atomicAdd(cntSrc + si, 1.0f);
        atomicAdd(cntDst + di, 1.0f);
    }

    const float* featRow = feat_emb + (long long)di * D;  // msg   A cols 0..63
    const float* nodeRow = node_emb + (long long)si * D;  // msg_f A cols 0..63
    const float* edgeRow = edge_emb + (long long)e * D;   // A cols 64..127 (shared)

    const v16h* Bf = (const v16h*)fragP;
    v8f acc[4]  = {v8f{}, v8f{}, v8f{}, v8f{}};
    v8f accf[4] = {v8f{}, v8f{}, v8f{}, v8f{}};

    #pragma unroll
    for (int c = 0; c < 2; ++c) {      // K = 0..63 : distinct A halves
        v16h aM = load_a_frag(featRow, c * 32, h, 1.0f);
        v16h aF = load_a_frag(nodeRow, c * 32, h, 1.0f);
        #pragma unroll
        for (int nt = 0; nt < 4; ++nt) {
            v16h b = Bf[(c * 4 + nt) * 32 + lane];
            acc[nt]  = wmma16(aM, b, acc[nt]);
            accf[nt] = wmma16(aF, b, accf[nt]);
        }
    }
    #pragma unroll
    for (int c = 2; c < 4; ++c) {      // K = 64..127 : shared edge_emb half
        v16h aE = load_a_frag(edgeRow, (c - 2) * 32, h, 1.0f);
        #pragma unroll
        for (int nt = 0; nt < 4; ++nt) {
            v16h b = Bf[(c * 4 + nt) * 32 + lane];
            acc[nt]  = wmma16(aE, b, acc[nt]);
            accf[nt] = wmma16(aE, b, accf[nt]);
        }
    }

    // Rows held by this lane: M = r + 8*h. Fetch their scatter indices.
    int srcRow[8], dstRow[8];
    #pragma unroll
    for (int r = 0; r < 8; ++r) {
        srcRow[r] = __shfl(si, r + 8 * h, 32);
        dstRow[r] = __shfl(di, r + 8 * h, 32);
    }

    #pragma unroll
    for (int nt = 0; nt < 4; ++nt) {
        int n = nt * 16 + m;
        float bias = bP[n];
        #pragma unroll
        for (int r = 0; r < 8; ++r) {
            float v = acc[nt][r] + bias;
            v = v > 0.0f ? v : 0.0f;
            atomicAdd(msgSum + (long long)srcRow[r] * D + n, v);
            float vf = accf[nt][r] + bias;
            vf = vf > 0.0f ? vf : 0.0f;
            atomicAdd(msgfSum + (long long)dstRow[r] * D + n, vf);
        }
    }
}

// ---------------------------------------------------------------------------
// Phase 2: out = concat(emb, sum/max(cnt,1)) @ Q + bQ  (16 rows per wave)
// ---------------------------------------------------------------------------
__global__ void update_kernel(const float* __restrict__ emb,
                              const float* __restrict__ sum,
                              const float* __restrict__ cnt,
                              const _Float16* __restrict__ fragQ,
                              const float* __restrict__ bQ,
                              float* __restrict__ out,
                              int nRows)
{
    int wave = (blockIdx.x * blockDim.x + threadIdx.x) >> 5;
    int lane = threadIdx.x & 31;
    int r0 = wave * 16;
    if (r0 >= nRows) return;
    int h = lane >> 4;
    int m = lane & 15;
    int row = r0 + m;

    float rinv = 1.0f / fmaxf(cnt[row], 1.0f);
    const float* embRow = emb + (long long)row * D;
    const float* sumRow = sum + (long long)row * D;

    const v16h* Bf = (const v16h*)fragQ;
    v8f acc[4] = {v8f{}, v8f{}, v8f{}, v8f{}};

    #pragma unroll
    for (int c = 0; c < 2; ++c) {
        v16h a = load_a_frag(embRow, c * 32, h, 1.0f);
        #pragma unroll
        for (int nt = 0; nt < 4; ++nt)
            acc[nt] = wmma16(a, Bf[(c * 4 + nt) * 32 + lane], acc[nt]);
    }
    #pragma unroll
    for (int c = 2; c < 4; ++c) {      // mean applied during f16 conversion
        v16h a = load_a_frag(sumRow, (c - 2) * 32, h, rinv);
        #pragma unroll
        for (int nt = 0; nt < 4; ++nt)
            acc[nt] = wmma16(a, Bf[(c * 4 + nt) * 32 + lane], acc[nt]);
    }

    #pragma unroll
    for (int nt = 0; nt < 4; ++nt) {
        int n = nt * 16 + m;
        float bias = bQ[n];
        #pragma unroll
        for (int r = 0; r < 8; ++r)
            out[(long long)(r0 + r + 8 * h) * D + n] = acc[nt][r] + bias;
    }
}

// ---------------------------------------------------------------------------
// Phase 3: edge_out = concat(edge_emb, node_out[src], feat_out[dst]) @ W + bW
// K = 192 -> 6 chunks, 24 WMMAs per wave of 16 edges.
// ---------------------------------------------------------------------------
__global__ void edge_out_kernel(const float* __restrict__ edge_emb,
                                const float* __restrict__ node_out,
                                const float* __restrict__ feat_out,
                                const int* __restrict__ src,
                                const int* __restrict__ dst,
                                const _Float16* __restrict__ fragW,
                                const float* __restrict__ bW,
                                float* __restrict__ out,
                                int nEdges)
{
    int wave = (blockIdx.x * blockDim.x + threadIdx.x) >> 5;
    int lane = threadIdx.x & 31;
    int e0 = wave * 16;
    if (e0 >= nEdges) return;
    int h = lane >> 4;
    int m = lane & 15;
    int e = e0 + m;

    int si = src[e];
    int di = dst[e];
    const float* seg0 = edge_emb + (long long)e * D;
    const float* seg1 = node_out + (long long)si * D;
    const float* seg2 = feat_out + (long long)di * D;

    const v16h* Bf = (const v16h*)fragW;
    v8f acc[4] = {v8f{}, v8f{}, v8f{}, v8f{}};

    const float* segs[3] = {seg0, seg1, seg2};
    #pragma unroll
    for (int s = 0; s < 3; ++s) {
        #pragma unroll
        for (int c = 0; c < 2; ++c) {
            v16h a = load_a_frag(segs[s], c * 32, h, 1.0f);
            int chunk = s * 2 + c;
            #pragma unroll
            for (int nt = 0; nt < 4; ++nt)
                acc[nt] = wmma16(a, Bf[(chunk * 4 + nt) * 32 + lane], acc[nt]);
        }
    }

    #pragma unroll
    for (int nt = 0; nt < 4; ++nt) {
        int n = nt * 16 + m;
        float bias = bW[n];
        #pragma unroll
        for (int r = 0; r < 8; ++r)
            out[(long long)(e0 + r + 8 * h) * D + n] = acc[nt][r] + bias;
    }
}

// ---------------------------------------------------------------------------
// Launch
// ---------------------------------------------------------------------------
extern "C" void kernel_launch(void* const* d_in, const int* in_sizes, int n_in,
                              void* d_out, int out_size, void* d_ws, size_t ws_size,
                              hipStream_t stream) {
    const float* node_emb = (const float*)d_in[0];
    const float* edge_emb = (const float*)d_in[1];
    const float* feat_emb = (const float*)d_in[2];
    const int*   eidx     = (const int*)d_in[3];
    const float* P        = (const float*)d_in[4];
    const float* bP       = (const float*)d_in[5];
    const float* Q        = (const float*)d_in[6];
    const float* bQ       = (const float*)d_in[7];
    const float* W        = (const float*)d_in[8];
    const float* bW       = (const float*)d_in[9];

    int nNodes = in_sizes[0] / D;
    int nEdges = in_sizes[1] / D;
    int nFeat  = in_sizes[2] / D;
    const int* src = eidx;
    const int* dst = eidx + nEdges;

    // ---- workspace layout (f32 accumulators then f16 weight fragments) ----
    float* ws      = (float*)d_ws;
    float* msgSum  = ws;                                  // nNodes * 64
    float* msgfSum = msgSum  + (size_t)nNodes * D;        // nFeat  * 64
    float* cntSrc  = msgfSum + (size_t)nFeat  * D;        // nNodes
    float* cntDst  = cntSrc  + nNodes;                    // nFeat
    size_t off = (size_t)((cntDst + nFeat) - ws);
    off = (off + 7) & ~(size_t)7;                         // 32B align for v16h
    _Float16* fragP = (_Float16*)(ws + off);              // 128*64 f16
    _Float16* fragQ = fragP + 128 * D;                    // 128*64 f16
    _Float16* fragW = fragQ + 128 * D;                    // 192*64 f16

    float* outNode = (float*)d_out;
    float* outEdge = outNode + (size_t)nNodes * D;
    float* outFeat = outEdge + (size_t)nEdges * D;

    // ---- zero accumulators (every call: graph replay safe) ----
    int zn = nNodes * D + nFeat * D + nNodes + nFeat;
    zero_kernel<<<(zn + 255) / 256, 256, 0, stream>>>(ws, zn);

    // ---- pack weights into WMMA fragment order ----
    pack_kernel<<<(128 * D + 255) / 256, 256, 0, stream>>>(P, fragP, 128);
    pack_kernel<<<(128 * D + 255) / 256, 256, 0, stream>>>(Q, fragQ, 128);
    pack_kernel<<<(192 * D + 255) / 256, 256, 0, stream>>>(W, fragW, 192);

    // ---- phase 1: edge messages + scatter sums + counts ----
    int tilesE = (nEdges + 15) / 16;
    msg_kernel<<<(tilesE * 32 + 255) / 256, 256, 0, stream>>>(
        node_emb, edge_emb, feat_emb, src, dst, fragP, bP,
        msgSum, msgfSum, cntSrc, cntDst, nEdges);

    // ---- phase 2: node / feature updates ----
    int tilesN = (nNodes + 15) / 16;
    update_kernel<<<(tilesN * 32 + 255) / 256, 256, 0, stream>>>(
        node_emb, msgSum, cntSrc, fragQ, bQ, outNode, nNodes);
    int tilesF = (nFeat + 15) / 16;
    update_kernel<<<(tilesF * 32 + 255) / 256, 256, 0, stream>>>(
        feat_emb, msgfSum, cntDst, fragQ, bQ, outFeat, nFeat);

    // ---- phase 3: edge update from refreshed endpoints ----
    edge_out_kernel<<<(tilesE * 32 + 255) / 256, 256, 0, stream>>>(
        edge_emb, outNode, outFeat, src, dst, fragW, bW, outEdge, nEdges);
}
